// DeepSeekV2MoEMLP_65824668778904
// MI455X (gfx1250) — matrix operand order repaired and checked by reading, not compiled, not run on hardware
//
#include <hip/hip_runtime.h>

#define T_TOKENS 2048
#define HID      2048
#define INTER    1408
#define NEXP     16
#define TOPK     4
#define SHINTER  2816
#define CAPACITY 768
#define LDS_K    40   // 32 + 8 bf16 pad per row

typedef __bf16 v16bf __attribute__((ext_vector_type(16)));
typedef float  v8f   __attribute__((ext_vector_type(8)));
typedef int    v4i   __attribute__((vector_size(16)));

union FragU { struct { uint4 lo, hi; } u; v16bf v; };

#if defined(__HIP_DEVICE_COMPILE__) && __has_builtin(__builtin_amdgcn_global_load_async_to_lds_b128)
#define HAVE_ASYNC 1
#else
#define HAVE_ASYNC 0
#endif

__device__ __forceinline__ void async_b128(const unsigned short* g, unsigned short* l) {
#if HAVE_ASYNC
  __builtin_amdgcn_global_load_async_to_lds_b128(
      (__attribute__((address_space(1))) v4i*)(g),
      (__attribute__((address_space(3))) v4i*)(l), 0, 0);
#else
  *reinterpret_cast<uint4*>(l) = *reinterpret_cast<const uint4*>(g);
#endif
}

template <int N>
__device__ __forceinline__ void wait_async() {
#if defined(__HIP_DEVICE_COMPILE__) && __has_builtin(__builtin_amdgcn_s_wait_asynccnt)
  __builtin_amdgcn_s_wait_asynccnt(N);
#elif defined(__HIP_DEVICE_COMPILE__)
  asm volatile("s_wait_asynccnt %0" :: "n"(N));
#endif
}

#if defined(__HIP_DEVICE_COMPILE__)
// CDNA5 has bf16-native VALU; v_cvt_pk_bf16_f32 does an RNE pack of two floats
// into one dword in a single instruction (a -> low half, b -> high half).
__device__ __forceinline__ unsigned pack_bf16(float a, float b) {
  unsigned r;
  asm("v_cvt_pk_bf16_f32 %0, %1, %2" : "=v"(r) : "v"(a), "v"(b));
  return r;
}
#else
__device__ __forceinline__ unsigned pack_bf16(float a, float b) {
  unsigned ua = __float_as_uint(a);
  unsigned ub = __float_as_uint(b);
  ua += 0x7FFFu + ((ua >> 16) & 1u);
  ub += 0x7FFFu + ((ub >> 16) & 1u);
  return (ua >> 16) | (ub & 0xFFFF0000u);
}
#endif

__device__ __forceinline__ unsigned short f2bf(float a) {
  return (unsigned short)(pack_bf16(a, 0.f) & 0xFFFFu);
}

// A fragment: 16x32 bf16 tile, LDS row-major (stride LDS_K).
__device__ __forceinline__ v16bf ld_fragA(const unsigned short* base, int lane) {
  const unsigned short* p = base + (lane & 15) * LDS_K + ((lane >> 4) << 3);
  FragU f;
  f.u.lo = *reinterpret_cast<const uint4*>(p);
  f.u.hi = *reinterpret_cast<const uint4*>(p + 16);
  return f.v;
}

// B fragment: 32x16 tile stored transposed in LDS as [n][k] (stride LDS_K).
__device__ __forceinline__ v16bf ld_fragB(const unsigned short* base, int lane) {
  const unsigned short* p = base + (lane & 15) * LDS_K + ((lane >> 4) << 3);
  FragU f;
  f.u.lo = *reinterpret_cast<const uint4*>(p);
  f.u.hi = *reinterpret_cast<const uint4*>(p + 16);
  return f.v;
}

// Issue async copy of one 128x32 bf16 A tile into LDS (2 x b128 per thread).
__device__ __forceinline__ void issue_A(const unsigned short* Abase, unsigned short* Asl,
                                        int mbase, int lda, int k0, int tid) {
  #pragma unroll
  for (int it = 0; it < 2; ++it) {
    int idx = tid + it * 256;
    int row = idx >> 2;
    int c4 = (idx & 3) << 3;
    async_b128(Abase + (long long)(mbase + row) * lda + k0 + c4,
               Asl + row * LDS_K + c4);
  }
}

// ---------------- small kernels ----------------

__global__ void init_kernel(int* cnt) {
  if (threadIdx.x < NEXP) cnt[threadIdx.x] = 0;
}

__global__ __launch_bounds__(256) void cvt_kernel(const float* __restrict__ x,
                                                  unsigned short* __restrict__ xb, int n4) {
  int i = blockIdx.x * 256 + threadIdx.x;
  if (i >= n4) return;
  float4 v = reinterpret_cast<const float4*>(x)[i];
  uint2 o;
  o.x = pack_bf16(v.x, v.y);
  o.y = pack_bf16(v.z, v.w);
  reinterpret_cast<uint2*>(xb)[i] = o;
}

__global__ __launch_bounds__(256) void router_kernel(const float* __restrict__ x,
                                                     const float* __restrict__ wg,
                                                     int* __restrict__ topi,
                                                     float* __restrict__ topw) {
  int t = blockIdx.x;
  int tid = threadIdx.x;
  int e = tid >> 4;
  int l = tid & 15;
  const float* xp = x + (long long)t * HID;
  const float* wp = wg + (long long)e * HID;
  float p = 0.f;
  for (int i = l; i < HID; i += 16) p += xp[i] * wp[i];
  __shared__ float red[NEXP][17];
  __shared__ float logit[NEXP];
  red[e][l] = p;
  __syncthreads();
  if (l == 0) {
    float s = 0.f;
    #pragma unroll
    for (int i = 0; i < 16; ++i) s += red[e][i];
    logit[e] = s;
  }
  __syncthreads();
  if (tid == 0) {
    float m = logit[0];
    #pragma unroll
    for (int i = 1; i < NEXP; ++i) m = fmaxf(m, logit[i]);
    float pr[NEXP];
    float s = 0.f;
    #pragma unroll
    for (int i = 0; i < NEXP; ++i) { pr[i] = __expf(logit[i] - m); s += pr[i]; }
    float inv = 1.f / s;
    #pragma unroll
    for (int i = 0; i < NEXP; ++i) pr[i] *= inv;
    for (int k = 0; k < TOPK; ++k) {
      int bi = 0; float bv = -1.f;
      #pragma unroll
      for (int i = 0; i < NEXP; ++i) if (pr[i] > bv) { bv = pr[i]; bi = i; }
      topi[t * TOPK + k] = bi;
      topw[t * TOPK + k] = bv;   // ROUTED_SCALE == 1
      pr[bi] = -2.f;
    }
  }
}

__global__ void assign_kernel(const int* __restrict__ topi, int* __restrict__ spos,
                              int* __restrict__ cnt) {
  int t = blockIdx.x * blockDim.x + threadIdx.x;
  if (t >= T_TOKENS) return;
  for (int k = 0; k < TOPK; ++k) {
    int e = topi[t * TOPK + k];
    int pos = atomicAdd(&cnt[e], 1);
    spos[t * TOPK + k] = pos;
  }
}

__global__ __launch_bounds__(256) void fill_kernel(const unsigned short* __restrict__ xb,
                                                   const int* __restrict__ topi,
                                                   const int* __restrict__ spos,
                                                   unsigned short* __restrict__ buf) {
  int slot = blockIdx.x;
  int pos = spos[slot];
  if (pos >= CAPACITY) return;
  int e = topi[slot];
  int t = slot >> 2;
  const uint4* src = reinterpret_cast<const uint4*>(xb + (long long)t * HID);
  uint4* dst = reinterpret_cast<uint4*>(buf + ((long long)e * CAPACITY + pos) * HID);
  dst[threadIdx.x] = src[threadIdx.x];   // 256 * 16B = HID * 2B
}

// ---------------- fused dual-GEMM + SwiGLU (up projections) ----------------
// Tile: BM=128, BN=64, BK=32, double-buffered. 8 waves as 4(m) x 2(n),
// each wave computes a 32x32 region of BOTH G and U.

__global__ __launch_bounds__(256) void dual_swiglu_kernel(
    const unsigned short* __restrict__ A, long long sAE, int lda,
    const float* __restrict__ B1, const float* __restrict__ B2, long long sBE, int ldb,
    unsigned short* __restrict__ Hout, long long sHE, int ldh,
    const int* __restrict__ cnt, int Mcap, int N, int K) {
  const int e = blockIdx.z;
  const int mbase = blockIdx.y * 128;
  const int nbase = blockIdx.x * 64;
  int valid = Mcap;
  if (cnt) { int c = cnt[e]; valid = c < Mcap ? c : Mcap; }
  if (mbase >= valid) return;

  A += (long long)e * sAE;
  B1 += (long long)e * sBE;
  B2 += (long long)e * sBE;
  Hout += (long long)e * sHE;

  __shared__ alignas(16) unsigned short As[2 * 128 * LDS_K];
  __shared__ alignas(16) unsigned short B1s[2 * 64 * LDS_K];
  __shared__ alignas(16) unsigned short B2s[2 * 64 * LDS_K];

  const int tid = threadIdx.x;
  const int lane = tid & 31;
  const int wave = tid >> 5;
  const int wm = wave >> 1;  // 0..3 -> rows wm*32
  const int wn = wave & 1;   // 0..1 -> cols wn*32

  v8f accg[2][2], accu[2][2];
  #pragma unroll
  for (int s = 0; s < 2; ++s)
    #pragma unroll
    for (int t = 0; t < 2; ++t)
      #pragma unroll
      for (int i = 0; i < 8; ++i) { accg[s][t][i] = 0.f; accu[s][t][i] = 0.f; }

  const int nk = K >> 5;
  float b1a[4], b1b[4], b2a[4], b2b[4];

  // prologue: tile 0 in flight
  issue_A(A, As, mbase, lda, 0, tid);
  #pragma unroll
  for (int it = 0; it < 4; ++it) {
    int w = tid + it * 256;
    int n = w & 63;
    int kp = w >> 6;
    long long off = (long long)(2 * kp) * ldb + (nbase + n);
    b1a[it] = B1[off]; b1b[it] = B1[off + ldb];
    b2a[it] = B2[off]; b2b[it] = B2[off + ldb];
  }

  for (int kt = 0; kt < nk; ++kt) {
    const int cur = kt & 1;
    unsigned short* Asl = As + cur * (128 * LDS_K);
    unsigned short* B1l = B1s + cur * (64 * LDS_K);
    unsigned short* B2l = B2s + cur * (64 * LDS_K);

    // commit staged B regs for this tile (compiler waits loadcnt here)
    #pragma unroll
    for (int it = 0; it < 4; ++it) {
      int w = tid + it * 256;
      int n = w & 63;
      int kp = w >> 6;
      *reinterpret_cast<unsigned*>(&B1l[n * LDS_K + 2 * kp]) = pack_bf16(b1a[it], b1b[it]);
      *reinterpret_cast<unsigned*>(&B2l[n * LDS_K + 2 * kp]) = pack_bf16(b2a[it], b2b[it]);
    }

    const bool more = (kt + 1 < nk);
    if (more) {
      const int k0n = (kt + 1) << 5;
      issue_A(A, As + (cur ^ 1) * (128 * LDS_K), mbase, lda, k0n, tid);
      #pragma unroll
      for (int it = 0; it < 4; ++it) {
        int w = tid + it * 256;
        int n = w & 63;
        int kp = w >> 6;
        long long off = (long long)(k0n + 2 * kp) * ldb + (nbase + n);
        b1a[it] = B1[off]; b1b[it] = B1[off + ldb];
        b2a[it] = B2[off]; b2b[it] = B2[off + ldb];
      }
      wait_async<2>();   // current tile's 2 async copies done, next 2 may remain in flight
    } else {
      wait_async<0>();
    }
    __syncthreads();

    v16bf af[2], bg[2], bu[2];
    #pragma unroll
    for (int s = 0; s < 2; ++s) af[s] = ld_fragA(&Asl[(wm * 32 + s * 16) * LDS_K], lane);
    #pragma unroll
    for (int t = 0; t < 2; ++t) {
      bg[t] = ld_fragB(&B1l[(wn * 32 + t * 16) * LDS_K], lane);
      bu[t] = ld_fragB(&B2l[(wn * 32 + t * 16) * LDS_K], lane);
    }
    #pragma unroll
    for (int s = 0; s < 2; ++s)
      #pragma unroll
      for (int t = 0; t < 2; ++t) {
        accg[s][t] = __builtin_amdgcn_wmma_f32_16x16x32_bf16(
            false, af[s], false, bg[t], (short)0, accg[s][t], false, false);
        accu[s][t] = __builtin_amdgcn_wmma_f32_16x16x32_bf16(
            false, af[s], false, bu[t], (short)0, accu[s][t], false, false);
      }
    __syncthreads();
  }

  const int rl = lane & 15;
  const int rh = (lane >> 4) << 3;
  #pragma unroll
  for (int s = 0; s < 2; ++s)
    #pragma unroll
    for (int t = 0; t < 2; ++t) {
      int row0 = mbase + wm * 32 + s * 16 + rh;
      int col = nbase + wn * 32 + t * 16 + rl;
      #pragma unroll
      for (int i = 0; i < 8; ++i) {
        float g = accg[s][t][i];
        float u = accu[s][t][i];
        float hv = (g / (1.f + __expf(-g))) * u;  // silu(g) * u
        Hout[(long long)(row0 + i) * ldh + col] = f2bf(hv);
      }
    }
}

// ---------------- down projection GEMM ----------------
// Tile: BM=128, BN=128, BK=32, double-buffered. 8 waves as 4(m) x 2(n), each 32x64.

__global__ __launch_bounds__(256) void down_gemm_kernel(
    const unsigned short* __restrict__ A, long long sAE, int lda,
    const float* __restrict__ B, long long sBE, int ldb,
    float* __restrict__ Out, long long sOE, int ldo,
    const int* __restrict__ cnt, int Mcap, int N, int K) {
  const int e = blockIdx.z;
  const int mbase = blockIdx.y * 128;
  const int nbase = blockIdx.x * 128;
  int valid = Mcap;
  if (cnt) { int c = cnt[e]; valid = c < Mcap ? c : Mcap; }
  if (mbase >= valid) return;

  A += (long long)e * sAE;
  B += (long long)e * sBE;
  Out += (long long)e * sOE;

  __shared__ alignas(16) unsigned short As[2 * 128 * LDS_K];
  __shared__ alignas(16) unsigned short Bs[2 * 128 * LDS_K];

  const int tid = threadIdx.x;
  const int lane = tid & 31;
  const int wave = tid >> 5;
  const int wm = wave >> 1;  // rows wm*32
  const int wn = wave & 1;   // cols wn*64

  v8f acc[2][4];
  #pragma unroll
  for (int s = 0; s < 2; ++s)
    #pragma unroll
    for (int t = 0; t < 4; ++t)
      #pragma unroll
      for (int i = 0; i < 8; ++i) acc[s][t][i] = 0.f;

  const int nk = K >> 5;
  float ba[8], bb[8];

  issue_A(A, As, mbase, lda, 0, tid);
  #pragma unroll
  for (int it = 0; it < 8; ++it) {
    int w = tid + it * 256;
    int n = w & 127;
    int kp = w >> 7;
    long long off = (long long)(2 * kp) * ldb + (nbase + n);
    ba[it] = B[off]; bb[it] = B[off + ldb];
  }

  for (int kt = 0; kt < nk; ++kt) {
    const int cur = kt & 1;
    unsigned short* Asl = As + cur * (128 * LDS_K);
    unsigned short* Bl = Bs + cur * (128 * LDS_K);

    #pragma unroll
    for (int it = 0; it < 8; ++it) {
      int w = tid + it * 256;
      int n = w & 127;
      int kp = w >> 7;
      *reinterpret_cast<unsigned*>(&Bl[n * LDS_K + 2 * kp]) = pack_bf16(ba[it], bb[it]);
    }

    const bool more = (kt + 1 < nk);
    if (more) {
      const int k0n = (kt + 1) << 5;
      issue_A(A, As + (cur ^ 1) * (128 * LDS_K), mbase, lda, k0n, tid);
      #pragma unroll
      for (int it = 0; it < 8; ++it) {
        int w = tid + it * 256;
        int n = w & 127;
        int kp = w >> 7;
        long long off = (long long)(k0n + 2 * kp) * ldb + (nbase + n);
        ba[it] = B[off]; bb[it] = B[off + ldb];
      }
      wait_async<2>();
    } else {
      wait_async<0>();
    }
    __syncthreads();

    v16bf af[2], bf[4];
    #pragma unroll
    for (int s = 0; s < 2; ++s) af[s] = ld_fragA(&Asl[(wm * 32 + s * 16) * LDS_K], lane);
    #pragma unroll
    for (int t = 0; t < 4; ++t) bf[t] = ld_fragB(&Bl[(wn * 64 + t * 16) * LDS_K], lane);
    #pragma unroll
    for (int s = 0; s < 2; ++s)
      #pragma unroll
      for (int t = 0; t < 4; ++t)
        acc[s][t] = __builtin_amdgcn_wmma_f32_16x16x32_bf16(
            false, af[s], false, bf[t], (short)0, acc[s][t], false, false);
    __syncthreads();
  }

  const int rl = lane & 15;
  const int rh = (lane >> 4) << 3;
  #pragma unroll
  for (int s = 0; s < 2; ++s)
    #pragma unroll
    for (int t = 0; t < 4; ++t) {
      int row0 = mbase + wm * 32 + s * 16 + rh;
      int col = nbase + wn * 64 + t * 16 + rl;
      #pragma unroll
      for (int i = 0; i < 8; ++i)
        Out[(long long)(row0 + i) * ldo + col] = acc[s][t][i];
    }
}

// ---------------- combine: out += sum_k w_k * O[e_k][pos_k] ----------------

__global__ __launch_bounds__(256) void combine_kernel(const float* __restrict__ O,
                                                      const int* __restrict__ topi,
                                                      const float* __restrict__ topw,
                                                      const int* __restrict__ spos,
                                                      float* __restrict__ out) {
  int t = blockIdx.x;
  for (int c = threadIdx.x; c < HID; c += 256) {
    float acc = out[(long long)t * HID + c];
    #pragma unroll
    for (int k = 0; k < TOPK; ++k) {
      int pos = spos[t * TOPK + k];
      if (pos < CAPACITY) {
        int e = topi[t * TOPK + k];
        acc += topw[t * TOPK + k] * O[((long long)e * CAPACITY + pos) * HID + c];
      }
    }
    out[(long long)t * HID + c] = acc;
  }
}

// ---------------- launch ----------------

extern "C" void kernel_launch(void* const* d_in, const int* in_sizes, int n_in,
                              void* d_out, int out_size, void* d_ws, size_t ws_size,
                              hipStream_t stream) {
  (void)in_sizes; (void)n_in; (void)out_size; (void)ws_size;
  const float* x   = (const float*)d_in[0];
  const float* wg  = (const float*)d_in[1];
  const float* w1  = (const float*)d_in[2];
  const float* w3  = (const float*)d_in[3];
  const float* w2  = (const float*)d_in[4];
  const float* ws1 = (const float*)d_in[5];
  const float* ws3 = (const float*)d_in[6];
  const float* ws2 = (const float*)d_in[7];
  float* out = (float*)d_out;

  char* ws = (char*)d_ws;
  size_t cur = 0;
  auto alloc = [&](size_t bytes) -> void* {
    void* p = ws + cur;
    cur = (cur + bytes + 255) & ~(size_t)255;
    return p;
  };
  int*            cnt  = (int*)alloc(NEXP * sizeof(int));
  int*            topi = (int*)alloc((size_t)T_TOKENS * TOPK * sizeof(int));
  float*          topw = (float*)alloc((size_t)T_TOKENS * TOPK * sizeof(float));
  int*            spos = (int*)alloc((size_t)T_TOKENS * TOPK * sizeof(int));
  unsigned short* xb   = (unsigned short*)alloc((size_t)T_TOKENS * HID * 2);
  unsigned short* buf  = (unsigned short*)alloc((size_t)NEXP * CAPACITY * HID * 2);
  unsigned short* hbuf = (unsigned short*)alloc((size_t)NEXP * CAPACITY * INTER * 2);
  unsigned short* hs   = (unsigned short*)alloc((size_t)T_TOKENS * SHINTER * 2);
  float*          obuf = (float*)alloc((size_t)NEXP * CAPACITY * HID * 4);

  init_kernel<<<1, 32, 0, stream>>>(cnt);
  cvt_kernel<<<(T_TOKENS * HID / 4 + 255) / 256, 256, 0, stream>>>(x, xb, T_TOKENS * HID / 4);
  router_kernel<<<T_TOKENS, 256, 0, stream>>>(x, wg, topi, topw);
  assign_kernel<<<(T_TOKENS + 255) / 256, 256, 0, stream>>>(topi, spos, cnt);
  fill_kernel<<<T_TOKENS * TOPK, 256, 0, stream>>>(xb, topi, spos, buf);

  // routed up: h = silu(buf@w1)*(buf@w3)    [E, CAP, INTER]
  dual_swiglu_kernel<<<dim3(INTER / 64, CAPACITY / 128, NEXP), 256, 0, stream>>>(
      buf, (long long)CAPACITY * HID, HID,
      w1, w3, (long long)HID * INTER, INTER,
      hbuf, (long long)CAPACITY * INTER, INTER,
      cnt, CAPACITY, INTER, HID);

  // shared up: hs = silu(x@ws1)*(x@ws3)     [T, SHINTER]
  dual_swiglu_kernel<<<dim3(SHINTER / 64, T_TOKENS / 128, 1), 256, 0, stream>>>(
      xb, 0, HID,
      ws1, ws3, 0, SHINTER,
      hs, 0, SHINTER,
      nullptr, T_TOKENS, SHINTER, HID);

  // routed down: o = h @ w2                 [E, CAP, HID]
  down_gemm_kernel<<<dim3(HID / 128, CAPACITY / 128, NEXP), 256, 0, stream>>>(
      hbuf, (long long)CAPACITY * INTER, INTER,
      w2, (long long)INTER * HID, HID,
      obuf, (long long)CAPACITY * HID, HID,
      cnt, CAPACITY, HID, INTER);

  // shared down: out = hs @ ws2 (full overwrite of d_out)
  down_gemm_kernel<<<dim3(HID / 128, T_TOKENS / 128, 1), 256, 0, stream>>>(
      hs, 0, SHINTER,
      ws2, 0, HID,
      out, 0, HID,
      nullptr, T_TOKENS, HID, SHINTER);

  // out += weighted routed contributions
  combine_kernel<<<T_TOKENS, 256, 0, stream>>>(obuf, topi, topw, spos, out);
}
